// hypercorre_topk2_58205396796095
// MI455X (gfx1250) — matrix operand order repaired and checked by reading, not compiled, not run on hardware
//
#include <hip/hip_runtime.h>
#include <hip/hip_bf16.h>

typedef __attribute__((ext_vector_type(16))) __bf16 v16bf;
typedef __attribute__((ext_vector_type(8)))  float  v8f;

#define BB    2
#define TT    3
#define CC    256
#define NTOK  3600     // 60*60
#define NTOKP 3616     // padded to a multiple of 32 columns (one extra 16-tile)
#define CQ    256
#define KTOP  128
#define NTILES (NTOK / 16)         // 225
#define NPAIRS ((NTILES + 1) / 2)  // 113 (pair 112's second tile is padding)

// round-to-nearest-even fp32 -> bf16 (bit manipulation only; no __bf16 arith)
__device__ __forceinline__ unsigned short f2bf(float f) {
    unsigned int u = __float_as_uint(f);
    unsigned int r = (u + 0x7FFFu + ((u >> 16) & 1u)) >> 16;
    return (unsigned short)r;
}

// ---------------------------------------------------------------------------
// Projection: out[img, tok, d] = bf16( sum_c X[img, c, tok] * W[c, d] + bias[d] )
// X is channel-major [NIMG, C, NTOK]; classic LDS-tiled fp32 GEMM (minor cost).
// Output rows padded to NTOKP tokens per image (pad region left untouched).
// ---------------------------------------------------------------------------
__global__ __launch_bounds__(256) void proj_kernel(
    const float* __restrict__ X, const float* __restrict__ W,
    const float* __restrict__ bias, unsigned short* __restrict__ out)
{
    __shared__ float xs[32][17];
    __shared__ float wsh[32][17];
    const int tx  = threadIdx.x;          // d within tile
    const int ty  = threadIdx.y;          // token within tile
    const int tid = ty * 16 + tx;
    const int n0  = blockIdx.x * 16;
    const int d0  = blockIdx.y * 16;
    const int img = blockIdx.z;

    float acc = 0.0f;
    for (int c0 = 0; c0 < CC; c0 += 32) {
        for (int e = tid; e < 32 * 16; e += 256) {
            int c = e >> 4, t = e & 15;
            xs[c][t]  = X[((size_t)img * CC + c0 + c) * NTOK + n0 + t];
            wsh[c][t] = W[(size_t)(c0 + c) * CQ + d0 + t];
        }
        __syncthreads();
        #pragma unroll
        for (int c = 0; c < 32; ++c) acc += xs[c][ty] * wsh[c][tx];
        __syncthreads();
    }
    acc += bias[d0 + tx];
    out[((size_t)img * NTOKP + n0 + ty) * CQ + d0 + tx] = f2bf(acc);
}

// ---------------------------------------------------------------------------
// Correlation volume: atten[b,t,ns,nq] = scale * dot(k[b,ns,:], q[b,t,nq,:])
// bf16 WMMA 16x16x32, fp32 accumulation, K=CQ=256 in 8 steps.
// Per wave: 16(ns) x 32(nq) x 3T = 6 accumulators; A-frag reused 6x per step.
// Branch-free K-loop (q_proj is column-padded); explicit double-buffering of
// A/B fragments pipelines the 14 b128 loads under the 6 WMMAs of each step.
// ---------------------------------------------------------------------------
union Frag { v16bf v; uint4 q[2]; };

__global__ __launch_bounds__(256) void corr_wmma_kernel(
    const unsigned short* __restrict__ kp,   // [B, NTOKP, CQ] bf16 (padded)
    const unsigned short* __restrict__ qp,   // [B*T, NTOKP, CQ] bf16 (padded)
    float* __restrict__ atten)               // [B, T, NTOK, NTOK]
{
    const int lane    = threadIdx.x & 31;
    const int wave    = threadIdx.x >> 5;
    const int ns_tile = blockIdx.x;           // 0..224
    const int pair    = blockIdx.y * 8 + wave;
    const int b       = blockIdx.z;
    // Force wave-uniform (scalar) control flow so WMMA regions are never
    // EXEC-masked: EXEC must be all-ones for WMMA.
    if (__builtin_amdgcn_readfirstlane(pair >= NPAIRS ? 1 : 0)) return;

    const int ns0   = ns_tile * 16;
    const int nq0   = pair * 32;
    const int l15   = lane & 15;
    const int khalf = lane >> 4;

    // A 16x32 bf16: lane holds row l15, K = ks + khalf*16 .. +15 (32 bytes).
    const unsigned short* aBase = kp + ((size_t)b * NTOKP + ns0 + l15) * CQ;
    // B 32x16 bf16: lane holds col l15, K = ks + khalf*16 .. +15 (32 bytes).
    const unsigned short* bBase[TT];
    #pragma unroll
    for (int t = 0; t < TT; ++t)
        bBase[t] = qp + ((size_t)(b * TT + t) * NTOKP + nq0 + l15) * CQ + khalf * 16;

    Frag aF[2];
    Frag bF[2][TT * 2];
    v8f  acc[TT * 2] = {};

    auto loadA = [&](Frag& f, int ks) {
        f.q[0] = *(const uint4*)(aBase + ks + khalf * 16);
        f.q[1] = *(const uint4*)(aBase + ks + khalf * 16 + 8);
    };
    auto loadB = [&](Frag* fr, int ks) {
        #pragma unroll
        for (int t = 0; t < TT; ++t) {
            const unsigned short* p0 = bBase[t] + ks;
            const unsigned short* p1 = p0 + 16 * CQ;     // next 16 columns
            fr[t * 2 + 0].q[0] = *(const uint4*)(p0);
            fr[t * 2 + 0].q[1] = *(const uint4*)(p0 + 8);
            fr[t * 2 + 1].q[0] = *(const uint4*)(p1);
            fr[t * 2 + 1].q[1] = *(const uint4*)(p1 + 8);
        }
    };

    loadA(aF[0], 0);
    loadB(bF[0], 0);
    #pragma unroll
    for (int s = 0; s < CQ / 32; ++s) {
        const int cur = s & 1, nxt = cur ^ 1;
        if (s < CQ / 32 - 1) {            // prefetch next K-step (unconditional code)
            loadA(aF[nxt], (s + 1) * 32);
            loadB(bF[nxt], (s + 1) * 32);
        }
        #pragma unroll
        for (int t = 0; t < TT; ++t) {
            acc[t * 2 + 0] = __builtin_amdgcn_wmma_f32_16x16x32_bf16(
                false, aF[cur].v, false, bF[cur][t * 2 + 0].v,
                (short)0, acc[t * 2 + 0], false, false);
            acc[t * 2 + 1] = __builtin_amdgcn_wmma_f32_16x16x32_bf16(
                false, aF[cur].v, false, bF[cur][t * 2 + 1].v,
                (short)0, acc[t * 2 + 1], false, false);
        }
    }

    // D 16x16 f32: VGPR v -> row = v + khalf*8, col = l15. Nontemporal stores:
    // 311 MB stream must not evict the L2-resident q_proj working set.
    // Only the epilogue checks whether the second column tile is real.
    const bool col1ok =
        __builtin_amdgcn_readfirstlane((nq0 + 16) < NTOK ? 1 : 0) != 0;
    const float scale = 0.0625f;  // 256^-0.5
    #pragma unroll
    for (int v = 0; v < 8; ++v) {
        const size_t row = (size_t)(ns0 + khalf * 8 + v);
        const size_t col = (size_t)(nq0 + l15);
        #pragma unroll
        for (int t = 0; t < TT; ++t) {
            float* p = atten + (((size_t)b * TT + t) * NTOK + row) * NTOK + col;
            __builtin_nontemporal_store(acc[t * 2 + 0][v] * scale, p);
            if (col1ok)
                __builtin_nontemporal_store(acc[t * 2 + 1][v] * scale, p + 16);
        }
    }
}

// ---------------------------------------------------------------------------
// Top-k over the T-summed correlation row (sum fused in; never materialized).
// One 256-thread block per (b, ns); iterative argmax with index tie-breaking
// (smallest index wins, matching jax.lax.top_k, descending output order).
// ---------------------------------------------------------------------------
__global__ __launch_bounds__(256) void topk_kernel(
    const float* __restrict__ atten, int* __restrict__ idx_out)
{
    __shared__ float s[NTOK];
    __shared__ float rv[256];
    __shared__ int   ri[256];
    const int tid = threadIdx.x;
    const int ns  = blockIdx.x;
    const int b   = blockIdx.y;

    const float* a0 = atten + (((size_t)b * TT + 0) * NTOK + ns) * NTOK;
    const float* a1 = a0 + (size_t)NTOK * NTOK;
    const float* a2 = a1 + (size_t)NTOK * NTOK;
    for (int i = tid; i < NTOK; i += 256)
        s[i] = __builtin_nontemporal_load(a0 + i) +
               __builtin_nontemporal_load(a1 + i) +
               __builtin_nontemporal_load(a2 + i);
    __syncthreads();

    for (int j = 0; j < KTOP; ++j) {
        float bv = -3.4e38f; int bi = NTOK;
        for (int i = tid; i < NTOK; i += 256) {
            float v = s[i];
            if (v > bv) { bv = v; bi = i; }   // ascending scan keeps smallest idx on ties
        }
        rv[tid] = bv; ri[tid] = bi;
        __syncthreads();
        for (int st = 128; st > 0; st >>= 1) {
            if (tid < st) {
                float ov = rv[tid + st]; int oi = ri[tid + st];
                if (ov > rv[tid] || (ov == rv[tid] && oi < ri[tid])) {
                    rv[tid] = ov; ri[tid] = oi;
                }
            }
            __syncthreads();
        }
        if (tid == 0) {
            idx_out[((size_t)b * NTOK + ns) * KTOP + j] = ri[0];
            s[ri[0]] = -3.4e38f;
        }
        __syncthreads();
    }
}

// ---------------------------------------------------------------------------
// Gather selected scores for each T and softmax jointly over T*k per pixel.
// One 128-thread block per (b, ns); thread j owns top-k slot j (3 values).
// ---------------------------------------------------------------------------
__global__ __launch_bounds__(KTOP) void gather_softmax_kernel(
    const float* __restrict__ atten, const int* __restrict__ idx,
    float* __restrict__ out)
{
    __shared__ float red[KTOP];
    const int j  = threadIdx.x;
    const int ns = blockIdx.x;
    const int b  = blockIdx.y;
    const int id = idx[((size_t)b * NTOK + ns) * KTOP + j];

    float v[TT];
    #pragma unroll
    for (int t = 0; t < TT; ++t)
        v[t] = atten[(((size_t)b * TT + t) * NTOK + ns) * NTOK + id];

    float m = fmaxf(fmaxf(v[0], v[1]), v[2]);
    red[j] = m; __syncthreads();
    for (int st = KTOP / 2; st > 0; st >>= 1) {
        if (j < st) red[j] = fmaxf(red[j], red[j + st]);
        __syncthreads();
    }
    m = red[0];
    __syncthreads();

    float e[TT]; float ls = 0.0f;
    #pragma unroll
    for (int t = 0; t < TT; ++t) { e[t] = __expf(v[t] - m); ls += e[t]; }
    red[j] = ls; __syncthreads();
    for (int st = KTOP / 2; st > 0; st >>= 1) {
        if (j < st) red[j] += red[j + st];
        __syncthreads();
    }
    const float inv = 1.0f / red[0];
    #pragma unroll
    for (int t = 0; t < TT; ++t)
        out[(((size_t)b * TT + t) * NTOK + ns) * KTOP + j] = e[t] * inv;
}

// ---------------------------------------------------------------------------
extern "C" void kernel_launch(void* const* d_in, const int* in_sizes, int n_in,
                              void* d_out, int out_size, void* d_ws, size_t ws_size,
                              hipStream_t stream) {
    const float* query = (const float*)d_in[0];   // [B,T,C,H,W]
    const float* supp  = (const float*)d_in[1];   // [B,1,C,H,W]
    const float* Wq    = (const float*)d_in[2];   // [C,CQ]
    const float* bq    = (const float*)d_in[3];   // [CQ]
    const float* Wk    = (const float*)d_in[4];   // [C,CQ]
    const float* bk    = (const float*)d_in[5];   // [CQ]
    (void)in_sizes; (void)n_in; (void)out_size; (void)ws_size;  // topk=128 hardcoded

    char* ws = (char*)d_ws;
    size_t off = 0;
    unsigned short* qp = (unsigned short*)(ws + off);   // [B*T, NTOKP, CQ] bf16
    off += (size_t)BB * TT * NTOKP * CQ * sizeof(unsigned short);
    off = (off + 255) & ~(size_t)255;
    unsigned short* kp = (unsigned short*)(ws + off);   // [B, NTOKP, CQ] bf16
    off += (size_t)BB * NTOKP * CQ * sizeof(unsigned short);
    off = (off + 255) & ~(size_t)255;
    float* atten = (float*)(ws + off);            // [B,T,NTOK,NTOK] ~311 MB

    float* attn_out = (float*)d_out;                               // [B,T,n,k]
    int*   idx_out  = (int*)(attn_out + (size_t)BB * TT * NTOK * KTOP);  // [B,n,k]

    dim3 pblk(16, 16);
    proj_kernel<<<dim3(NTILES, CQ / 16, BB * TT), pblk, 0, stream>>>(query, Wq, bq, qp);
    proj_kernel<<<dim3(NTILES, CQ / 16, BB),      pblk, 0, stream>>>(supp,  Wk, bk, kp);

    corr_wmma_kernel<<<dim3(NTILES, (NPAIRS + 7) / 8, BB), 256, 0, stream>>>(kp, qp, atten);

    topk_kernel<<<dim3(NTOK, BB), 256, 0, stream>>>(atten, idx_out);
    gather_softmax_kernel<<<dim3(NTOK, BB), KTOP, 0, stream>>>(atten, idx_out, attn_out);
}
